// GraphConvLayer_75179107549718
// MI455X (gfx1250) — matrix operand order repaired and checked by reading, not compiled, not run on hardware
//
#include <hip/hip_runtime.h>
#include <hip/hip_bf16.h>

// ---------------------------------------------------------------------------
// GNN fused layer for MI455X (gfx1250, wave32, WMMA).
// GEMMs use v_wmma_f32_16x16x32_bf16 (A/B converted fp32->bf16 RNE, f32 accum),
// double-buffered b128 fragment loads, branch-free inner loop (rows clamped
// for loads, bounds enforced only at the store epilogue).
// Scatter/segment ops use f32 global atomics.
// ---------------------------------------------------------------------------

#define N_NODES 20000
#define N_EDGES 160000
#define DIM     256
#define NHEAD   4
#define DHEAD   64
#define NBATCH  64
#define EPSV    1e-5f

typedef __bf16 bf16_t;
typedef __attribute__((ext_vector_type(16))) __bf16 v16bf;
typedef __attribute__((ext_vector_type(8)))  float  v8f;

union FragBf { uint4 u[2]; v16bf v; };

__device__ inline bf16_t f2bf(float f) {
  unsigned int u = __float_as_uint(f);
  u += 0x7FFFu + ((u >> 16) & 1u);                 // round-to-nearest-even
  unsigned short s = (unsigned short)(u >> 16);
  return __builtin_bit_cast(bf16_t, s);
}

__device__ inline void atomicMaxF(float* addr, float val) {
  unsigned int* ua = (unsigned int*)addr;
  unsigned int old = *ua;
  while (__uint_as_float(old) < val) {
    unsigned int assumed = old;
    old = atomicCAS(ua, assumed, __float_as_uint(val));
    if (old == assumed) break;
  }
}

__device__ inline float waveReduceSum(float v) {
  #pragma unroll
  for (int o = 16; o > 0; o >>= 1) v += __shfl_xor(v, o, 32);
  return v;
}

__device__ inline float blockReduceSum256(float v) {
  __shared__ float s[8];
  __shared__ float r;
  v = waveReduceSum(v);
  int lane = threadIdx.x & 31, w = threadIdx.x >> 5;
  if (lane == 0) s[w] = v;
  __syncthreads();
  if (threadIdx.x == 0) {
    float t = 0.f;
    #pragma unroll
    for (int i = 0; i < 8; ++i) t += s[i];
    r = t;
  }
  __syncthreads();
  float out = r;
  __syncthreads();   // allow safe reuse of s/r by a second call
  return out;
}

// ---------------------------------------------------------------------------
// Tiled bf16 WMMA GEMM: C[M x Nt] = A[M x K](bf16,row-major, lda) *
//   Bt[Nt x K](bf16, weight pre-transposed, row n at Bt[n*ldb + boff]) + bias.
// Block = 256 threads = 8 waves; block tile 128(M) x 128(N); wave tile 32x64
// (2x4 WMMA 16x16x32 fragments), double-buffered over K (K % 64 == 0).
// Nt must be a multiple of 128 (true for all call sites: 256 / 1024).
// ---------------------------------------------------------------------------
__global__ __launch_bounds__(256)
void gemm_bf16_kernel(const bf16_t* __restrict__ A, int lda,
                      const bf16_t* __restrict__ Bt, int ldb, int boff,
                      const float* __restrict__ bias,
                      float* __restrict__ C, int ldc,
                      int M, int K, int act)
{
  const int lane  = threadIdx.x & 31;
  const int wave  = threadIdx.x >> 5;
  const int wm    = wave & 3;                 // 4 waves along M
  const int wn    = wave >> 2;                // 2 waves along N
  const int row0  = blockIdx.x * 128 + wm * 32;
  const int col0  = blockIdx.y * 128 + wn * 64;
  const int laneR = lane & 15;
  const int laneH = lane >> 4;                // 0/1 half-wave
  const int kA    = laneH * 8;                // A: K sub-chunk (ISA 16-bit A layout)
  const int kB    = laneH * 8;                // B: K sub-chunk (same layout as A)

  // Clamp A rows for loads (always in-bounds); stores are bounds-checked.
  int rA0 = row0 + laneR;      if (rA0 > M - 1) rA0 = M - 1;
  int rA1 = row0 + 16 + laneR; if (rA1 > M - 1) rA1 = M - 1;
  const bf16_t* pA[2] = { A + (size_t)rA0 * lda + kA,
                          A + (size_t)rA1 * lda + kA };
  const bf16_t* pB[4];
  #pragma unroll
  for (int j = 0; j < 4; ++j)
    pB[j] = Bt + (size_t)(col0 + j * 16 + laneR) * ldb + boff + kB;

  v8f acc[2][4];
  #pragma unroll
  for (int i = 0; i < 2; ++i)
    #pragma unroll
    for (int j = 0; j < 4; ++j)
      #pragma unroll
      for (int g = 0; g < 8; ++g) acc[i][j][g] = 0.f;

  FragBf aP[2], bP[4], aQ[2], bQ[4];

  auto loadFrags = [&](FragBf* a, FragBf* b, int k0) {
    #pragma unroll
    for (int i = 0; i < 2; ++i) {
      a[i].u[0] = *(const uint4*)(pA[i] + k0);        // K = k0+kA    .. +7
      a[i].u[1] = *(const uint4*)(pA[i] + k0 + 16);   // K = k0+16+kA .. +7
    }
    #pragma unroll
    for (int j = 0; j < 4; ++j) {
      b[j].u[0] = *(const uint4*)(pB[j] + k0);        // K = k0+kB    .. +7
      b[j].u[1] = *(const uint4*)(pB[j] + k0 + 16);   // K = k0+16+kB .. +7
    }
  };
  auto mma = [&](FragBf* a, FragBf* b) {
    #pragma unroll
    for (int i = 0; i < 2; ++i)
      #pragma unroll
      for (int j = 0; j < 4; ++j)
        acc[i][j] = __builtin_amdgcn_wmma_f32_16x16x32_bf16(
            false, a[i].v, false, b[j].v, (short)0, acc[i][j], false, false);
  };

  loadFrags(aP, bP, 0);
  for (int k0 = 0; k0 < K; k0 += 64) {
    loadFrags(aQ, bQ, k0 + 32);                 // next half in flight
    mma(aP, bP);
    int kn = (k0 + 64 < K) ? (k0 + 64) : 0;     // branch-free dummy reload on last
    loadFrags(aP, bP, kn);
    mma(aQ, bQ);
  }

  // D layout: VGPR g -> (M = laneH*8 + g, N = laneR) within each 16x16 tile.
  #pragma unroll
  for (int i = 0; i < 2; ++i) {
    int rbase = row0 + i * 16 + laneH * 8;
    #pragma unroll
    for (int j = 0; j < 4; ++j) {
      int c = col0 + j * 16 + laneR;
      float bv = bias ? bias[c] : 0.f;
      #pragma unroll
      for (int g = 0; g < 8; ++g) {
        int r = rbase + g;
        if (r < M) {
          float x = acc[i][j][g] + bv;
          if (act == 1) x = fmaxf(x, 0.f);
          C[(size_t)r * ldc + c] = x;
        }
      }
    }
  }
}

// --------------------------- utility kernels -------------------------------
__global__ __launch_bounds__(256)
void fill_kernel(float* __restrict__ p, long n, float v) {
  long i = (long)blockIdx.x * 256 + threadIdx.x;
  if (i < n) p[i] = v;
}

// fp32 -> bf16 (optional per-column bias, colMask = pow2-1)
__global__ __launch_bounds__(256)
void cvt_bf16_kernel(const float* __restrict__ in, const float* __restrict__ bias,
                     int colMask, bf16_t* __restrict__ out, long n) {
  long i = (long)blockIdx.x * 256 + threadIdx.x;
  if (i >= n) return;
  float v = in[i];
  if (bias) v += bias[i & colMask];
  out[i] = f2bf(v);
}

// W[K][Nc] fp32 -> Wt[Nc][K] bf16
__global__ __launch_bounds__(256)
void transpose_w_kernel(const float* __restrict__ W, bf16_t* __restrict__ Wt,
                        int K, int Nc) {
  long i = (long)blockIdx.x * 256 + threadIdx.x;
  if (i >= (long)K * Nc) return;
  int k = (int)(i / Nc), n = (int)(i % Nc);
  Wt[(size_t)n * K + k] = f2bf(W[i]);
}

// ------------------------- TransformerConv branch --------------------------
__global__ __launch_bounds__(256)
void attn_logits_kernel(const float* __restrict__ q, const float* __restrict__ k,
                        const float* __restrict__ ep, const int* __restrict__ src,
                        const int* __restrict__ dst, float* __restrict__ logits) {
  long gw = ((long)blockIdx.x * 256 + threadIdx.x) >> 5;
  if (gw >= (long)N_EDGES * NHEAD) return;
  int lane = threadIdx.x & 31;
  int e = (int)(gw >> 2), h = (int)(gw & 3);
  int s = src[e], d = dst[e];
  const float* qp = q  + (size_t)d * DIM + h * DHEAD;
  const float* kp = k  + (size_t)s * DIM + h * DHEAD;
  const float* pp = ep + (size_t)e * DIM + h * DHEAD;
  float sum = 0.f;
  #pragma unroll
  for (int i = 0; i < 2; ++i) {
    int idx = lane + i * 32;
    sum = fmaf(qp[idx], kp[idx] + pp[idx], sum);
  }
  sum = waveReduceSum(sum);
  if (lane == 0) logits[gw] = sum * 0.125f;   // 1/sqrt(64)
}

__global__ __launch_bounds__(256)
void segmax_kernel(const float* __restrict__ logits, const int* __restrict__ dst,
                   float* __restrict__ segmax, long n) {
  long i = (long)blockIdx.x * 256 + threadIdx.x;
  if (i >= n) return;
  int e = (int)(i >> 2), h = (int)(i & 3);
  atomicMaxF(&segmax[(size_t)dst[e] * NHEAD + h], logits[i]);
}

__global__ __launch_bounds__(256)
void expsum_kernel(const float* __restrict__ logits, const int* __restrict__ dst,
                   const float* __restrict__ segmax, float* __restrict__ acoef,
                   float* __restrict__ segsum, long n) {
  long i = (long)blockIdx.x * 256 + threadIdx.x;
  if (i >= n) return;
  int e = (int)(i >> 2), h = (int)(i & 3);
  float x = __expf(logits[i] - segmax[(size_t)dst[e] * NHEAD + h]);
  acoef[i] = x;
  atomicAdd(&segsum[(size_t)dst[e] * NHEAD + h], x);
}

__global__ __launch_bounds__(256)
void attn_accum_kernel(const float* __restrict__ acoef, const float* __restrict__ segsum,
                       const float* __restrict__ v, const float* __restrict__ ep,
                       const int* __restrict__ src, const int* __restrict__ dst,
                       float* __restrict__ out) {
  long i = (long)blockIdx.x * 256 + threadIdx.x;
  if (i >= (long)N_EDGES * DIM) return;
  int e = (int)(i >> 8), dd = (int)(i & 255), h = dd >> 6;
  int s = src[e], d = dst[e];
  float coef = acoef[(size_t)e * NHEAD + h] /
               (segsum[(size_t)d * NHEAD + h] + 1e-16f);
  atomicAdd(&out[(size_t)d * DIM + dd],
            coef * (v[(size_t)s * DIM + dd] + ep[i]));
}

__global__ __launch_bounds__(256)
void residual_ln_kernel(const float* __restrict__ base, const float* __restrict__ add,
                        const float* __restrict__ g, const float* __restrict__ b,
                        float* __restrict__ out) {
  int r = blockIdx.x, t = threadIdx.x;
  float x = base[(size_t)r * DIM + t] + add[(size_t)r * DIM + t];
  float m = blockReduceSum256(x) * (1.f / DIM);
  float dx = x - m;
  float var = blockReduceSum256(dx * dx) * (1.f / DIM);
  out[(size_t)r * DIM + t] = dx * rsqrtf(var + EPSV) * g[t] + b[t];
}

// ------------------------------ GAT branch ---------------------------------
__global__ __launch_bounds__(256)
void gat_att_kernel(const float* __restrict__ xg, const float* __restrict__ att_s,
                    const float* __restrict__ att_d, float* __restrict__ asrc,
                    float* __restrict__ adst) {
  long gw = ((long)blockIdx.x * 256 + threadIdx.x) >> 5;
  if (gw >= (long)N_NODES * NHEAD) return;
  int lane = threadIdx.x & 31;
  int n = (int)(gw >> 2), h = (int)(gw & 3);
  const float* xp = xg + (size_t)n * (NHEAD * DIM) + h * DIM;
  const float* s1 = att_s + h * DIM;
  const float* s2 = att_d + h * DIM;
  float ss = 0.f, sd = 0.f;
  for (int i = lane; i < DIM; i += 32) {
    float v = xp[i];
    ss = fmaf(v, s1[i], ss);
    sd = fmaf(v, s2[i], sd);
  }
  ss = waveReduceSum(ss);
  sd = waveReduceSum(sd);
  if (lane == 0) { asrc[gw] = ss; adst[gw] = sd; }
}

__global__ __launch_bounds__(256)
void gat_logits_kernel(const float* __restrict__ asrc, const float* __restrict__ adst,
                       const int* __restrict__ src, const int* __restrict__ dst,
                       float* __restrict__ logits, long n) {
  long i = (long)blockIdx.x * 256 + threadIdx.x;
  if (i >= n) return;
  int e = (int)(i >> 2), h = (int)(i & 3);
  float x = asrc[(size_t)src[e] * NHEAD + h] + adst[(size_t)dst[e] * NHEAD + h];
  logits[i] = (x > 0.f) ? x : 0.2f * x;    // leaky_relu(0.2)
}

__global__ __launch_bounds__(256)
void gat_accum_kernel(const float* __restrict__ acoef, const float* __restrict__ segsum,
                      const float* __restrict__ xg, const int* __restrict__ src,
                      const int* __restrict__ dst, float* __restrict__ out) {
  long i = (long)blockIdx.x * 256 + threadIdx.x;
  if (i >= (long)N_EDGES * (NHEAD * DIM)) return;
  int e = (int)(i >> 10), dd = (int)(i & 1023), h = dd >> 8;
  int s = src[e], d = dst[e];
  float coef = acoef[(size_t)e * NHEAD + h] /
               (segsum[(size_t)d * NHEAD + h] + 1e-16f);
  atomicAdd(&out[(size_t)d * (NHEAD * DIM) + dd],
            coef * xg[(size_t)s * (NHEAD * DIM) + dd]);
}

// ------------------------------ GCN branch ---------------------------------
__global__ __launch_bounds__(256)
void deg_kernel(const int* __restrict__ dst, float* __restrict__ deg) {
  long i = (long)blockIdx.x * 256 + threadIdx.x;
  if (i < N_EDGES) atomicAdd(&deg[dst[i]], 1.f);
}

__global__ __launch_bounds__(256)
void gcn_accum_kernel(const float* __restrict__ xw, const float* __restrict__ deg,
                      const int* __restrict__ src, const int* __restrict__ dst,
                      float* __restrict__ out) {
  long i = (long)blockIdx.x * 256 + threadIdx.x;
  if (i >= (long)N_EDGES * DIM) return;
  int e = (int)(i >> 8), dd = (int)(i & 255);
  int s = src[e], d = dst[e];
  float nrm = rsqrtf(deg[s]) * rsqrtf(deg[d]);
  atomicAdd(&out[(size_t)d * DIM + dd], nrm * xw[(size_t)s * DIM + dd]);
}

__global__ __launch_bounds__(256)
void gcn_finish_kernel(float* __restrict__ acc, const float* __restrict__ xw,
                       const float* __restrict__ deg, const float* __restrict__ bias) {
  long i = (long)blockIdx.x * 256 + threadIdx.x;
  if (i >= (long)N_NODES * DIM) return;
  int r = (int)(i >> 8), c = (int)(i & 255);
  acc[i] = acc[i] + xw[i] / deg[r] + bias[c];
}

// ----------------------------- fusion / FFN --------------------------------
__global__ __launch_bounds__(256)
void fuse_kernel(const float* __restrict__ gat, const float* __restrict__ gcn,
                 const float* __restrict__ trn, const float* __restrict__ fw,
                 float* __restrict__ out) {
  long i = (long)blockIdx.x * 256 + threadIdx.x;
  if (i >= (long)N_NODES * DIM) return;
  float a = fw[0], b = fw[1], c = fw[2];
  float m = fmaxf(a, fmaxf(b, c));
  float ea = __expf(a - m), eb = __expf(b - m), ec = __expf(c - m);
  float inv = 1.f / (ea + eb + ec);
  out[i] = (ea * gat[i] + eb * gcn[i] + ec * trn[i]) * inv;
}

// ------------------------------ edge update --------------------------------
__global__ __launch_bounds__(256)
void edge_hidden_kernel(const float* __restrict__ P, const float* __restrict__ R,
                        const int* __restrict__ src, const int* __restrict__ dst,
                        float* __restrict__ Q /* in-place: Q += P[src]+R[dst]; gelu */) {
  long i = (long)blockIdx.x * 256 + threadIdx.x;
  if (i >= (long)N_EDGES * DIM) return;
  int e = (int)(i >> 8), c = (int)(i & 255);
  float x = P[(size_t)src[e] * DIM + c] + Q[i] + R[(size_t)dst[e] * DIM + c];
  Q[i] = 0.5f * x * (1.f + erff(x * 0.70710678118654752f));   // exact gelu
}

__global__ __launch_bounds__(256)
void edge_residual_kernel(float* __restrict__ msg, const float* __restrict__ ef) {
  long i = (long)blockIdx.x * 256 + threadIdx.x;
  if (i >= (long)N_EDGES * DIM) return;
  msg[i] = ef[i] + msg[i];
}

__global__ __launch_bounds__(256)
void bn_stats_kernel(const float* __restrict__ er, float* __restrict__ sums /*[512]*/) {
  int t = threadIdx.x;
  float s = 0.f, q = 0.f;
  for (int r = blockIdx.x; r < N_EDGES; r += gridDim.x) {
    float v = er[(size_t)r * DIM + t];
    s += v;
    q = fmaf(v, v, q);
  }
  atomicAdd(&sums[t], s);
  atomicAdd(&sums[DIM + t], q);
}

__global__ __launch_bounds__(256)
void bn_apply_kernel(const float* __restrict__ er, const float* __restrict__ sums,
                     const float* __restrict__ g, const float* __restrict__ b,
                     float* __restrict__ out) {
  long i = (long)blockIdx.x * 256 + threadIdx.x;
  if (i >= (long)N_EDGES * DIM) return;
  int c = (int)(i & 255);
  const float invE = 1.f / (float)N_EDGES;
  float m = sums[c] * invE;
  float var = sums[DIM + c] * invE - m * m;
  out[i] = (er[i] - m) * rsqrtf(var + EPSV) * g[c] + b[c];
}

// --------------------------- node context gate -----------------------------
__global__ __launch_bounds__(256)
void seg_count_kernel(const int* __restrict__ batch, float* __restrict__ cnt) {
  long i = (long)blockIdx.x * 256 + threadIdx.x;
  if (i < N_NODES) atomicAdd(&cnt[batch[i]], 1.f);
}

__global__ __launch_bounds__(256)
void seg_accum_kernel(const float* __restrict__ x, const int* __restrict__ batch,
                      float* __restrict__ mean) {
  long i = (long)blockIdx.x * 256 + threadIdx.x;
  if (i >= (long)N_NODES * DIM) return;
  int n = (int)(i >> 8), c = (int)(i & 255);
  atomicAdd(&mean[(size_t)batch[n] * DIM + c], x[i]);
}

__global__ __launch_bounds__(256)
void seg_div_kernel(float* __restrict__ mean, const float* __restrict__ cnt) {
  long i = (long)blockIdx.x * 256 + threadIdx.x;
  if (i >= (long)NBATCH * DIM) return;
  mean[i] /= fmaxf(cnt[i >> 8], 1.f);
}

__global__ __launch_bounds__(256)
void nc_gate_kernel(const float* __restrict__ mean,
                    const float* __restrict__ W1, const float* __restrict__ b1,
                    const float* __restrict__ W2, const float* __restrict__ b2,
                    float* __restrict__ gate) {
  __shared__ float hid[DIM];
  int b = blockIdx.x, t = threadIdx.x;
  const float* mr = mean + (size_t)b * DIM;
  float acc = b1[t];
  for (int k = 0; k < DIM; ++k) acc = fmaf(mr[k], W1[(size_t)k * DIM + t], acc);
  hid[t] = fmaxf(acc, 0.f);
  __syncthreads();
  float acc2 = b2[t];
  for (int k = 0; k < DIM; ++k) acc2 = fmaf(hid[k], W2[(size_t)k * DIM + t], acc2);
  gate[(size_t)b * DIM + t] = 1.f / (1.f + __expf(-acc2));
}

__global__ __launch_bounds__(256)
void apply_gate_kernel(const float* __restrict__ x, const float* __restrict__ gate,
                       const int* __restrict__ batch, float* __restrict__ out) {
  long i = (long)blockIdx.x * 256 + threadIdx.x;
  if (i >= (long)N_NODES * DIM) return;
  int n = (int)(i >> 8), c = (int)(i & 255);
  out[i] = x[i] * gate[(size_t)batch[n] * DIM + c];
}

// ---------------------------------------------------------------------------
extern "C" void kernel_launch(void* const* d_in, const int* in_sizes, int n_in,
                              void* d_out, int out_size, void* d_ws, size_t ws_size,
                              hipStream_t stream) {
  (void)in_sizes; (void)n_in; (void)out_size;

  const float* node_feat = (const float*)d_in[0];
  const int*   links     = (const int*)d_in[1];
  const float* edge_feat = (const float*)d_in[2];
  const int*   batch     = (const int*)d_in[3];
  const float* tc_Wq = (const float*)d_in[4];  const float* tc_bq = (const float*)d_in[5];
  const float* tc_Wk = (const float*)d_in[6];  const float* tc_bk = (const float*)d_in[7];
  const float* tc_Wv = (const float*)d_in[8];  const float* tc_bv = (const float*)d_in[9];
  const float* tc_We = (const float*)d_in[10];
  const float* gat_W = (const float*)d_in[11];
  const float* att_s = (const float*)d_in[12]; const float* att_d = (const float*)d_in[13];
  const float* gat_b = (const float*)d_in[14];
  const float* adp_W = (const float*)d_in[15]; const float* adp_b = (const float*)d_in[16];
  const float* gcn_W = (const float*)d_in[17]; const float* gcn_b = (const float*)d_in[18];
  const float* ln_g  = (const float*)d_in[19]; const float* ln_b  = (const float*)d_in[20];
  const float* fw    = (const float*)d_in[21];
  const float* ffn_W1 = (const float*)d_in[22]; const float* ffn_b1 = (const float*)d_in[23];
  const float* ffn_W2 = (const float*)d_in[24]; const float* ffn_b2 = (const float*)d_in[25];
  const float* eu_W1 = (const float*)d_in[26]; const float* eu_b1 = (const float*)d_in[27];
  const float* eu_W2 = (const float*)d_in[28]; const float* eu_b2 = (const float*)d_in[29];
  const float* eu_g  = (const float*)d_in[30]; const float* eu_be = (const float*)d_in[31];
  const float* nc_W1 = (const float*)d_in[32]; const float* nc_b1 = (const float*)d_in[33];
  const float* nc_W2 = (const float*)d_in[34]; const float* nc_b2 = (const float*)d_in[35];

  const int* src = links;
  const int* dst = links + N_EDGES;

  const long ND  = (long)N_NODES * DIM;
  const long ED  = (long)N_EDGES * DIM;
  const long N4D = (long)N_NODES * NHEAD * DIM;
  const long E4D = (long)N_EDGES * NHEAD * DIM;
  const long EH  = (long)N_EDGES * NHEAD;
  const long NHh = (long)N_NODES * NHEAD;

  // ---------------- workspace bump allocator ----------------
  char* base = (char*)d_ws;
  size_t off = 0;
  auto alloc = [&](size_t bytes) -> void* {
    void* p = base + off;
    off += (bytes + 255) & ~(size_t)255;
    return p;
  };

  bf16_t* wqT   = (bf16_t*)alloc(DIM * DIM * 2);
  bf16_t* wkT   = (bf16_t*)alloc(DIM * DIM * 2);
  bf16_t* wvT   = (bf16_t*)alloc(DIM * DIM * 2);
  bf16_t* weT   = (bf16_t*)alloc(DIM * DIM * 2);
  bf16_t* gcnT  = (bf16_t*)alloc(DIM * DIM * 2);
  bf16_t* euW2T = (bf16_t*)alloc(DIM * DIM * 2);
  bf16_t* gatWT = (bf16_t*)alloc((size_t)DIM * NHEAD * DIM * 2);   // [1024][256]
  bf16_t* adpWT = (bf16_t*)alloc((size_t)DIM * NHEAD * DIM * 2);   // [256][1024]
  bf16_t* ffn1T = (bf16_t*)alloc((size_t)DIM * 4 * DIM * 2);       // [1024][256]
  bf16_t* ffn2T = (bf16_t*)alloc((size_t)DIM * 4 * DIM * 2);       // [256][1024]
  bf16_t* eu1T  = (bf16_t*)alloc((size_t)3 * DIM * DIM * 2);       // [256][768]
  bf16_t* nbf   = (bf16_t*)alloc(ND * 2);
  bf16_t* ebf   = (bf16_t*)alloc(ED * 2);
  float*  qb    = (float*)alloc(ND * 4);
  float*  kb    = (float*)alloc(ND * 4);
  float*  vb    = (float*)alloc(ND * 4);
  float*  R1    = (float*)alloc(ED * 4);          // eproj -> Qe -> eh -> msg -> er
  float*  logits= (float*)alloc(EH * 4);
  float*  acoef = (float*)alloc(EH * 4);
  float*  segmx = (float*)alloc(NHh * 4);
  float*  segsm = (float*)alloc(NHh * 4);
  float*  asrc  = (float*)alloc(NHh * 4);
  float*  adst  = (float*)alloc(NHh * 4);
  float*  tmpN  = (float*)alloc(ND * 4);          // trans_out -> gat_out -> gcn acc -> ffn out
  float*  trnR  = (float*)alloc(ND * 4);
  float*  R2    = (float*)alloc(N4D * 4);         // xg -> ffn hidden f32
  float*  R3f   = (float*)alloc(N4D * 4);         // gat accum; later reused as ehbf (E*D bf16)
  float*  gatR  = (float*)alloc(ND * 4);
  float*  xw    = (float*)alloc(ND * 4);
  float*  deg   = (float*)alloc(N_NODES * 4);
  float*  gcnR  = (float*)alloc(ND * 4);
  float*  fused = (float*)alloc(ND * 4);
  float*  node2 = (float*)alloc(ND * 4);
  float*  Pbuf  = (float*)alloc(ND * 4);
  float*  Rbuf  = (float*)alloc(ND * 4);
  bf16_t* smlbf = (bf16_t*)alloc(ND * 2);         // fused bf16 -> node2 bf16
  bf16_t* hidbf = (bf16_t*)alloc(N4D * 2);        // gat-concat bf16 -> ffn hidden bf16
  float*  bnsum = (float*)alloc(2 * DIM * 4);
  float*  meanB = (float*)alloc((size_t)NBATCH * DIM * 4);
  float*  cntB  = (float*)alloc(NBATCH * 4);
  float*  gateB = (float*)alloc((size_t)NBATCH * DIM * 4);
  bf16_t* ehbf  = (bf16_t*)R3f;                   // alias: gat accum region reused

  if (off > ws_size) return;   // workspace too small: do nothing (deterministic)

  float* out_node = (float*)d_out;
  float* out_edge = (float*)d_out + ND;

  auto blks = [](long n) { return dim3((unsigned)((n + 255) / 256)); };
  auto gemm = [&](const bf16_t* A, int lda, const bf16_t* Bt, int ldb, int boff,
                  const float* bias, float* C, int M, int Nt, int K, int act) {
    dim3 g((M + 127) / 128, Nt / 128);
    gemm_bf16_kernel<<<g, 256, 0, stream>>>(A, lda, Bt, ldb, boff, bias, C, Nt, M, K, act);
  };
  auto tr = [&](const float* W, bf16_t* Wt, int K, int Nc) {
    transpose_w_kernel<<<blks((long)K * Nc), 256, 0, stream>>>(W, Wt, K, Nc);
  };

  // -------- weight prep (fp32 -> bf16, transposed to [N][K]) --------
  tr(tc_Wq, wqT, DIM, DIM);       tr(tc_Wk, wkT, DIM, DIM);
  tr(tc_Wv, wvT, DIM, DIM);       tr(tc_We, weT, DIM, DIM);
  tr(gcn_W, gcnT, DIM, DIM);      tr(eu_W2, euW2T, DIM, DIM);
  tr(gat_W, gatWT, DIM, NHEAD * DIM);
  tr(adp_W, adpWT, NHEAD * DIM, DIM);
  tr(ffn_W1, ffn1T, DIM, 4 * DIM);
  tr(ffn_W2, ffn2T, 4 * DIM, DIM);
  tr(eu_W1, eu1T, 3 * DIM, DIM);
  cvt_bf16_kernel<<<blks(ND), 256, 0, stream>>>(node_feat, nullptr, 0, nbf, ND);
  cvt_bf16_kernel<<<blks(ED), 256, 0, stream>>>(edge_feat, nullptr, 0, ebf, ED);

  // -------- TransformerConv --------
  gemm(nbf, DIM, wqT, DIM, 0, tc_bq, qb, N_NODES, DIM, DIM, 0);
  gemm(nbf, DIM, wkT, DIM, 0, tc_bk, kb, N_NODES, DIM, DIM, 0);
  gemm(nbf, DIM, wvT, DIM, 0, tc_bv, vb, N_NODES, DIM, DIM, 0);
  gemm(ebf, DIM, weT, DIM, 0, nullptr, R1, N_EDGES, DIM, DIM, 0);
  fill_kernel<<<blks(NHh), 256, 0, stream>>>(segmx, NHh, -1e30f);
  fill_kernel<<<blks(NHh), 256, 0, stream>>>(segsm, NHh, 0.f);
  fill_kernel<<<blks(ND), 256, 0, stream>>>(tmpN, ND, 0.f);
  attn_logits_kernel<<<blks(EH * 32), 256, 0, stream>>>(qb, kb, R1, src, dst, logits);
  segmax_kernel<<<blks(EH), 256, 0, stream>>>(logits, dst, segmx, EH);
  expsum_kernel<<<blks(EH), 256, 0, stream>>>(logits, dst, segmx, acoef, segsm, EH);
  attn_accum_kernel<<<blks(ED), 256, 0, stream>>>(acoef, segsm, vb, R1, src, dst, tmpN);
  residual_ln_kernel<<<N_NODES, 256, 0, stream>>>(node_feat, tmpN, ln_g, ln_b, trnR);

  // -------- GAT --------
  gemm(nbf, DIM, gatWT, DIM, 0, nullptr, R2, N_NODES, NHEAD * DIM, DIM, 0);
  gat_att_kernel<<<blks(NHh * 32), 256, 0, stream>>>(R2, att_s, att_d, asrc, adst);
  fill_kernel<<<blks(NHh), 256, 0, stream>>>(segmx, NHh, -1e30f);
  fill_kernel<<<blks(NHh), 256, 0, stream>>>(segsm, NHh, 0.f);
  fill_kernel<<<blks(N4D), 256, 0, stream>>>(R3f, N4D, 0.f);
  gat_logits_kernel<<<blks(EH), 256, 0, stream>>>(asrc, adst, src, dst, logits, EH);
  segmax_kernel<<<blks(EH), 256, 0, stream>>>(logits, dst, segmx, EH);
  expsum_kernel<<<blks(EH), 256, 0, stream>>>(logits, dst, segmx, acoef, segsm, EH);
  gat_accum_kernel<<<blks(E4D), 256, 0, stream>>>(acoef, segsm, R2, src, dst, R3f);
  cvt_bf16_kernel<<<blks(N4D), 256, 0, stream>>>(R3f, gat_b, NHEAD * DIM - 1, hidbf, N4D);
  gemm(hidbf, NHEAD * DIM, adpWT, NHEAD * DIM, 0, adp_b, tmpN, N_NODES, DIM, NHEAD * DIM, 0);
  residual_ln_kernel<<<N_NODES, 256, 0, stream>>>(node_feat, tmpN, ln_g + DIM, ln_b + DIM, gatR);

  // -------- GCN --------
  gemm(nbf, DIM, gcnT, DIM, 0, nullptr, xw, N_NODES, DIM, DIM, 0);
  fill_kernel<<<blks(N_NODES), 256, 0, stream>>>(deg, N_NODES, 1.f);
  deg_kernel<<<blks(N_EDGES), 256, 0, stream>>>(dst, deg);
  fill_kernel<<<blks(ND), 256, 0, stream>>>(tmpN, ND, 0.f);
  gcn_accum_kernel<<<blks(ED), 256, 0, stream>>>(xw, deg, src, dst, tmpN);
  gcn_finish_kernel<<<blks(ND), 256, 0, stream>>>(tmpN, xw, deg, gcn_b);
  residual_ln_kernel<<<N_NODES, 256, 0, stream>>>(node_feat, tmpN, ln_g + 2 * DIM, ln_b + 2 * DIM, gcnR);

  // -------- fusion + FFN --------
  fuse_kernel<<<blks(ND), 256, 0, stream>>>(gatR, gcnR, trnR, fw, fused);
  cvt_bf16_kernel<<<blks(ND), 256, 0, stream>>>(fused, nullptr, 0, smlbf, ND);
  gemm(smlbf, DIM, ffn1T, DIM, 0, ffn_b1, R2, N_NODES, 4 * DIM, DIM, 1);     // relu
  cvt_bf16_kernel<<<blks(N4D), 256, 0, stream>>>(R2, nullptr, 0, hidbf, N4D);
  gemm(hidbf, 4 * DIM, ffn2T, 4 * DIM, 0, ffn_b2, tmpN, N_NODES, DIM, 4 * DIM, 0);
  residual_ln_kernel<<<N_NODES, 256, 0, stream>>>(fused, tmpN, ln_g + 3 * DIM, ln_b + 3 * DIM, node2);

  // -------- edge update (eu_W1 split: rows 0..255->src, 256..511->edge, 512..767->dst) --------
  cvt_bf16_kernel<<<blks(ND), 256, 0, stream>>>(node2, nullptr, 0, smlbf, ND);
  gemm(smlbf, DIM, eu1T, 3 * DIM, 0,       nullptr, Pbuf, N_NODES, DIM, DIM, 0);
  gemm(smlbf, DIM, eu1T, 3 * DIM, 2 * DIM, nullptr, Rbuf, N_NODES, DIM, DIM, 0);
  gemm(ebf,   DIM, eu1T, 3 * DIM, DIM,     eu_b1,   R1,   N_EDGES, DIM, DIM, 0);   // Qe
  edge_hidden_kernel<<<blks(ED), 256, 0, stream>>>(Pbuf, Rbuf, src, dst, R1);      // gelu in-place
  cvt_bf16_kernel<<<blks(ED), 256, 0, stream>>>(R1, nullptr, 0, ehbf, ED);
  gemm(ehbf, DIM, euW2T, DIM, 0, eu_b2, R1, N_EDGES, DIM, DIM, 0);                 // msg
  edge_residual_kernel<<<blks(ED), 256, 0, stream>>>(R1, edge_feat);               // er
  fill_kernel<<<blks(2 * DIM), 256, 0, stream>>>(bnsum, 2 * DIM, 0.f);
  bn_stats_kernel<<<256, 256, 0, stream>>>(R1, bnsum);
  bn_apply_kernel<<<blks(ED), 256, 0, stream>>>(R1, bnsum, eu_g, eu_be, out_edge);

  // -------- node context gate --------
  fill_kernel<<<blks((long)NBATCH * DIM), 256, 0, stream>>>(meanB, (long)NBATCH * DIM, 0.f);
  fill_kernel<<<blks(NBATCH), 256, 0, stream>>>(cntB, NBATCH, 0.f);
  seg_count_kernel<<<blks(N_NODES), 256, 0, stream>>>(batch, cntB);
  seg_accum_kernel<<<blks(ND), 256, 0, stream>>>(node2, batch, meanB);
  seg_div_kernel<<<blks((long)NBATCH * DIM), 256, 0, stream>>>(meanB, cntB);
  nc_gate_kernel<<<NBATCH, 256, 0, stream>>>(meanB, nc_W1, nc_b1, nc_W2, nc_b2, gateB);
  apply_gate_kernel<<<NBATCH ? blks(ND) : 1, 256, 0, stream>>>(node2, gateB, batch, out_node);
}
